// MultiViewGNN_73057393705161
// MI455X (gfx1250) — matrix operand order repaired and checked by reading, not compiled, not run on hardware
//
#include <hip/hip_runtime.h>
#include <hip/hip_bf16.h>

#define NNODES 50000
#define NEDGES 800000
#define FIN    256
#define HEADS  4
#define CDIM   128
#define HC     512            // HEADS*CDIM
#define ETOT   (NEDGES + NNODES)
#define MTILES 3125           // NNODES/16

typedef __attribute__((ext_vector_type(16))) __bf16       v16bf;
typedef __attribute__((ext_vector_type(8)))  float        v8f;
typedef __attribute__((ext_vector_type(4)))  unsigned int v4u;
typedef __attribute__((ext_vector_type(8)))  int          v8i;
typedef __attribute__((ext_vector_type(4)))  int          v4i;

// ---------------- zero fill (graph-capture-safe init) ----------------
__global__ void fill_zero_f32(float* __restrict__ p, int n) {
  int i = blockIdx.x * blockDim.x + threadIdx.x;
  if (i < n) p[i] = 0.f;
}

// ---------------- GEMM: h = x @ W  (M=50000, K=256, N=512) ----------------
// Block = 8 waves = 128 rows x 64 cols. W k-slab (256x64 f32, 64KB) staged into
// LDS once per block by the Tensor Data Mover; each wave computes 16x64 with one
// A fragment reused across 4 v_wmma_f32_16x16x32_bf16 per k-step.
__global__ void __launch_bounds__(256)
gemm_wmma(const float* __restrict__ x, const float* __restrict__ W,
          float* __restrict__ hbuf) {
  __shared__ float wlds[FIN * 64];            // [k][c] row-major, 64 KB

  const int lane = threadIdx.x & 31;
  const int wave = threadIdx.x >> 5;
  const int mb   = blockIdx.x >> 3;           // m-group (8 m-tiles per block)
  const int nb   = blockIdx.x & 7;            // n-group (64 cols)
  const float* __restrict__ wsl = W + nb * 64;  // slab origin (f32 elems)

#if __has_builtin(__builtin_amdgcn_tensor_load_to_lds)
  if (wave == 0) {
    // Build Tensor DMA Descriptor (D#), ISA cdna5 §8: 2-D tile 64 x 256,
    // data_size=4B, tensor_dim0_stride=512 (W row pitch).
    unsigned long long ga = (unsigned long long)(const char*)wsl;
    unsigned lds_base = (unsigned)(unsigned long long)(const void*)wlds; // addr[31:0] == LDS offset
    v4u g0;
    g0[0] = 1u;                                             // count=1 (valid user D#)
    g0[1] = lds_base;                                       // lds_addr
    g0[2] = (unsigned)(ga & 0xFFFFFFFFu);                   // global_addr[31:0]
    g0[3] = (unsigned)(((ga >> 32) & 0x01FFFFFFu) | (2u << 30)); // addr[56:32] | type=2
    v8i g1;
    g1[0] = (int)(2u << 16);          // data_size = 4 bytes
    g1[1] = (int)(64u << 16);         // tensor_dim0[15:0] = 64   (bits 79:48)
    g1[2] = (int)(256u << 16);        // td0 hi=0 | tensor_dim1[15:0] = 256
    g1[3] = (int)(64u << 16);         // td1 hi=0 | tile_dim0 = 64
    g1[4] = (int)256;                 // tile_dim1 = 256, tile_dim2 = 0
    g1[5] = (int)512;                 // tensor_dim0_stride[31:0] = 512
    g1[6] = 0;                        // stride0 hi | stride1 lo
    g1[7] = 0;
    v4i z4 = {0, 0, 0, 0};
#if defined(__clang_major__) && (__clang_major__ >= 23)
    v8i z8 = {0, 0, 0, 0, 0, 0, 0, 0};
    __builtin_amdgcn_tensor_load_to_lds(g0, g1, z4, z4, z8, 0);   // tensor_load_to_lds
#else
    __builtin_amdgcn_tensor_load_to_lds(g0, g1, z4, z4, 0);       // tensor_load_to_lds
#endif
    __builtin_amdgcn_s_wait_tensorcnt(0);                         // s_wait_tensorcnt 0
  }
#else
  // Fallback: cooperative coalesced copy, 4096 float4s.
  for (int idx = threadIdx.x; idx < FIN * 16; idx += 256) {
    int k = idx >> 4, c4 = idx & 15;
    ((float4*)wlds)[k * 16 + c4] = ((const float4*)(wsl + (size_t)k * HC))[c4];
  }
#endif
  __syncthreads();

  const int tm = mb * 8 + wave;               // this wave's m-tile
  if (tm >= MTILES) return;                   // (no barriers below)

  const int row  = tm * 16 + (lane & 15);     // A row this lane owns
  const int ksub = (lane >> 4) * 8;           // A K sub-offset (ISA 16-bit A layout)
  const int kb   = (lane >> 4) * 8;           // B K sub-offset (matches A table order)
  const int ncl  = lane & 15;                 // local column within 16-wide tile

  v8f acc[4] = {};
  for (int kk = 0; kk < FIN; kk += 32) {
    // ---- A fragment: 16 bf16 per lane from x (two 128b + two 128b loads)
    const float4* xv = (const float4*)(x + (size_t)row * FIN + kk + ksub);
    float4 a0 = xv[0], a1 = xv[1];            // K = ksub+0..7
    const float4* xw = (const float4*)(x + (size_t)row * FIN + kk + ksub + 16);
    float4 a2 = xw[0], a3 = xw[1];            // K = 16+ksub+0..7
    v16bf a;
    a[0]=(__bf16)a0.x;  a[1]=(__bf16)a0.y;  a[2]=(__bf16)a0.z;  a[3]=(__bf16)a0.w;
    a[4]=(__bf16)a1.x;  a[5]=(__bf16)a1.y;  a[6]=(__bf16)a1.z;  a[7]=(__bf16)a1.w;
    a[8]=(__bf16)a2.x;  a[9]=(__bf16)a2.y;  a[10]=(__bf16)a2.z; a[11]=(__bf16)a2.w;
    a[12]=(__bf16)a3.x; a[13]=(__bf16)a3.y; a[14]=(__bf16)a3.z; a[15]=(__bf16)a3.w;

    // ---- 4 B fragments from LDS, one WMMA each (A reused)
#pragma unroll
    for (int t = 0; t < 4; ++t) {
      const float* bl = wlds + (size_t)(kk + kb) * 64 + t * 16 + ncl;
      v16bf bm;
#pragma unroll
      for (int i = 0; i < 16; ++i)            // elem i -> K = kk+kb+i (+8 for i>=8)
        bm[i] = (__bf16)bl[(i + (i & 8)) * 64];
      acc[t] = __builtin_amdgcn_wmma_f32_16x16x32_bf16(
          false, a, false, bm, (short)0, acc[t], false, false);
    }
  }

  // ---- store D: VGPR j -> M = rbase+j, N = nb*64 + t*16 + ncl
  const int rbase = tm * 16 + (lane >> 4) * 8;
#pragma unroll
  for (int t = 0; t < 4; ++t) {
    float* out = hbuf + (size_t)rbase * HC + nb * 64 + t * 16 + ncl;
#pragma unroll
    for (int j = 0; j < 8; ++j)
      out[(size_t)j * HC] = acc[t][j];
  }
}

// ---------------- per-node attention dots: s_src[n,h], s_dst[n,h] ----------------
__global__ void sdots(const float* __restrict__ hbuf, const float* __restrict__ a_src,
                      const float* __restrict__ a_dst, float* __restrict__ s_src,
                      float* __restrict__ s_dst) {
  int t = blockIdx.x * blockDim.x + threadIdx.x;
  if (t >= NNODES * HEADS) return;
  int n = t >> 2, hh = t & 3;
  const float4* hp = (const float4*)(hbuf + (size_t)n * HC + hh * CDIM);
  const float4* as = (const float4*)(a_src + hh * CDIM);
  const float4* ad = (const float4*)(a_dst + hh * CDIM);
  float ss = 0.f, sd = 0.f;
#pragma unroll 8
  for (int i = 0; i < CDIM / 4; ++i) {
    float4 hv = hp[i], av = as[i], dv = ad[i];
    ss += hv.x * av.x + hv.y * av.y + hv.z * av.z + hv.w * av.w;
    sd += hv.x * dv.x + hv.y * dv.y + hv.z * dv.z + hv.w * dv.w;
  }
  s_src[t] = ss;
  s_dst[t] = sd;
}

// monotonic uint encoding of f32 so atomicMax(u32) == float max
__device__ __forceinline__ unsigned f2o(float f) {
  unsigned u = __float_as_uint(f);
  return (u >> 31) ? ~u : (u | 0x80000000u);
}
__device__ __forceinline__ float o2f(unsigned u) {
  return (u >> 31) ? __uint_as_float(u & 0x7FFFFFFFu) : __uint_as_float(~u);
}

// ---------------- pass A: e = leaky_relu(s_src[src]+s_dst[dst]); segment max ----------------
__global__ void edge_max(const int* __restrict__ ei, const float* __restrict__ s_src,
                         const float* __restrict__ s_dst, float* __restrict__ ew,
                         unsigned* __restrict__ mu) {
  int e = blockIdx.x * blockDim.x + threadIdx.x;
  if (e >= ETOT) return;
  int s, d;
  if (e < NEDGES) { s = ei[e]; d = ei[NEDGES + e]; }
  else            { s = e - NEDGES; d = s; }         // self loops
  float4 ev;
  float* evp = (float*)&ev;
#pragma unroll
  for (int h = 0; h < 4; ++h) {
    float v = s_src[s * 4 + h] + s_dst[d * 4 + h];
    v = (v > 0.f) ? v : 0.2f * v;
    evp[h] = v;
    atomicMax(mu + d * 4 + h, f2o(v));
  }
  ((float4*)ew)[e] = ev;
}

// ---------------- pass B: ex = exp(e - m[dst]); segment sum ----------------
__global__ void edge_exp(const int* __restrict__ ei, float* __restrict__ ew,
                         const unsigned* __restrict__ mu, float* __restrict__ denom) {
  int e = blockIdx.x * blockDim.x + threadIdx.x;
  if (e >= ETOT) return;
  int d = (e < NEDGES) ? ei[NEDGES + e] : (e - NEDGES);
  float4 ev = ((float4*)ew)[e];
  float* evp = (float*)&ev;
#pragma unroll
  for (int h = 0; h < 4; ++h) {
    float ex = __expf(evp[h] - o2f(mu[d * 4 + h]));
    evp[h] = ex;
    atomicAdd(denom + d * 4 + h, ex);
  }
  ((float4*)ew)[e] = ev;
}

// ---------------- pass C: acc[dst,c] += (1/H) * sum_h alpha_h * h[src,h,c] ----------------
// one wave per edge; lanes coalesced over channels; head-mean folded into alpha
__global__ void edge_aggregate(const int* __restrict__ ei, const float* __restrict__ ew,
                               const float* __restrict__ denom,
                               const float* __restrict__ hbuf, float* __restrict__ acc) {
  int gw   = (blockIdx.x * blockDim.x + threadIdx.x) >> 5;
  int lane = threadIdx.x & 31;
  if (gw >= ETOT) return;
  int s, d;
  if (gw < NEDGES) { s = ei[gw]; d = ei[NEDGES + gw]; }
  else             { s = gw - NEDGES; d = s; }
  const float* hp = hbuf + (size_t)s * HC;
  __builtin_prefetch(hp + lane * 16, 0, 3);          // global_prefetch_b8 on gathered row
  float al[4];
#pragma unroll
  for (int h = 0; h < 4; ++h)
    al[h] = 0.25f * ew[gw * 4 + h] / fmaxf(denom[d * 4 + h], 1e-16f);
  float* ad = acc + (size_t)d * CDIM;
#pragma unroll
  for (int cc = 0; cc < 4; ++cc) {
    int c = lane + cc * 32;
    float sum = al[0] * hp[c] + al[1] * hp[CDIM + c] +
                al[2] * hp[2 * CDIM + c] + al[3] * hp[3 * CDIM + c];
    atomicAdd(ad + c, sum);                          // global_atomic_add_f32, no-return
  }
}

// ---------------- view-attention scores + softmax weights ----------------
__global__ void fuse_scores(const float* __restrict__ acc, const float* __restrict__ b,
                            const float* __restrict__ fus_w, const float* __restrict__ fus_b,
                            float* __restrict__ wout) {
  int n = blockIdx.x * blockDim.x + threadIdx.x;
  if (n >= NNODES) return;
  float sc[3];
#pragma unroll
  for (int v = 0; v < 3; ++v) {
    const float* av = acc + ((size_t)v * NNODES + n) * CDIM;
    const float* bv = b + v * CDIM;
    float s = fus_b[0];
    for (int c = 0; c < CDIM; ++c) {
      float xx = av[c] + bv[c];
      xx = (xx > 0.f) ? xx : (__expf(xx) - 1.f);     // ELU
      s += xx * fus_w[c];
    }
    sc[v] = s;
  }
  float m  = fmaxf(sc[0], fmaxf(sc[1], sc[2]));
  float e0 = __expf(sc[0] - m), e1 = __expf(sc[1] - m), e2 = __expf(sc[2] - m);
  float inv = 1.f / (e0 + e1 + e2);
  wout[n * 3 + 0] = e0 * inv;
  wout[n * 3 + 1] = e1 * inv;
  wout[n * 3 + 2] = e2 * inv;
}

// ---------------- fused representation + 2-layer classifier ----------------
__global__ void __launch_bounds__(128)
classify(const float* __restrict__ acc, const float* __restrict__ b,
         const float* __restrict__ w1, const float* __restrict__ b1,
         const float* __restrict__ w2, const float* __restrict__ b2,
         const float* __restrict__ wview, float* __restrict__ logits) {
  __shared__ float w1s[CDIM * 64];   // 32 KB
  __shared__ float w2s[64 * 2];
  __shared__ float b1s[64];
  for (int i = threadIdx.x; i < CDIM * 64; i += blockDim.x) w1s[i] = w1[i];
  for (int i = threadIdx.x; i < 128;       i += blockDim.x) w2s[i] = w2[i];
  for (int i = threadIdx.x; i < 64;        i += blockDim.x) b1s[i] = b1[i];
  __syncthreads();
  int n = blockIdx.x * blockDim.x + threadIdx.x;
  if (n >= NNODES) return;
  float wv0 = wview[n * 3 + 0], wv1 = wview[n * 3 + 1], wv2 = wview[n * 3 + 2];
  const float* a0 = acc + (size_t)n * CDIM;
  const float* a1 = acc + ((size_t)NNODES     + n) * CDIM;
  const float* a2 = acc + ((size_t)2 * NNODES + n) * CDIM;
  float hcls[64];
#pragma unroll
  for (int j = 0; j < 64; ++j) hcls[j] = b1s[j];
  for (int c = 0; c < CDIM; ++c) {
    float x0 = a0[c] + b[c];            x0 = (x0 > 0.f) ? x0 : (__expf(x0) - 1.f);
    float x1 = a1[c] + b[CDIM + c];     x1 = (x1 > 0.f) ? x1 : (__expf(x1) - 1.f);
    float x2 = a2[c] + b[2 * CDIM + c]; x2 = (x2 > 0.f) ? x2 : (__expf(x2) - 1.f);
    float f = wv0 * x0 + wv1 * x1 + wv2 * x2;
#pragma unroll
    for (int j = 0; j < 64; ++j) hcls[j] += f * w1s[c * 64 + j];
  }
  float l0 = b2[0], l1 = b2[1];
#pragma unroll
  for (int j = 0; j < 64; ++j) {
    float s = fmaxf(hcls[j], 0.f);
    l0 += s * w2s[j * 2 + 0];
    l1 += s * w2s[j * 2 + 1];
  }
  logits[n * 2 + 0] = l0;
  logits[n * 2 + 1] = l1;
}

// ---------------- launcher ----------------
extern "C" void kernel_launch(void* const* d_in, const int* in_sizes, int n_in,
                              void* d_out, int out_size, void* d_ws, size_t ws_size,
                              hipStream_t stream) {
  const float* x     = (const float*)d_in[0];
  const int*   ei[3] = {(const int*)d_in[1], (const int*)d_in[2], (const int*)d_in[3]};
  const float* W     = (const float*)d_in[4];
  const float* a_src = (const float*)d_in[5];
  const float* a_dst = (const float*)d_in[6];
  const float* b     = (const float*)d_in[7];
  const float* fus_w = (const float*)d_in[8];
  const float* fus_b = (const float*)d_in[9];
  const float* w1    = (const float*)d_in[10];
  const float* b1    = (const float*)d_in[11];
  const float* w2    = (const float*)d_in[12];
  const float* b2    = (const float*)d_in[13];

  float* logits = (float*)d_out;              // [N,2]
  float* wview  = (float*)d_out + NNODES * 2; // [N,3]

  char* ws = (char*)d_ws;
  size_t off = 0;
  auto carve = [&](size_t bytes) -> void* {
    void* p = ws + off;
    off += (bytes + 255) & ~(size_t)255;
    return p;
  };
  float*    acc   = (float*)   carve(sizeof(float) * 3 * NNODES * CDIM);   // 76.8 MB
  float*    hbuf  = (float*)   carve(sizeof(float) * (size_t)NNODES * HC); // 102.4 MB
  float*    s_src = (float*)   carve(sizeof(float) * NNODES * HEADS);
  float*    s_dst = (float*)   carve(sizeof(float) * NNODES * HEADS);
  unsigned* mu    = (unsigned*)carve(sizeof(unsigned) * NNODES * HEADS);   // contiguous with denom
  float*    denom = (float*)   carve(sizeof(float) * NNODES * HEADS);
  float*    ew    = (float*)   carve(sizeof(float) * (size_t)ETOT * HEADS); // 13.6 MB
  (void)denom; // denom == mu + NNODES*HEADS (256-aligned carve, sizes are multiples of 256)

  {
    int ntot = 3 * NNODES * CDIM;
    fill_zero_f32<<<(ntot + 255) / 256, 256, 0, stream>>>(acc, ntot);
  }
  const int mgroups = (MTILES + 7) / 8;       // 391
  for (int v = 0; v < 3; ++v) {
    {
      int nz = NNODES * HEADS * 2;  // zero mu + denom in one pass (contiguous)
      fill_zero_f32<<<(nz + 255) / 256, 256, 0, stream>>>((float*)mu, nz);
    }
    gemm_wmma<<<mgroups * 8, 256, 0, stream>>>(x, W + (size_t)v * FIN * HC, hbuf);
    sdots<<<(NNODES * HEADS + 255) / 256, 256, 0, stream>>>(
        hbuf, a_src + v * HC, a_dst + v * HC, s_src, s_dst);
    edge_max<<<(ETOT + 255) / 256, 256, 0, stream>>>(ei[v], s_src, s_dst, ew, mu);
    edge_exp<<<(ETOT + 255) / 256, 256, 0, stream>>>(ei[v], ew, mu, denom);
    edge_aggregate<<<(ETOT + 7) / 8, 256, 0, stream>>>(
        ei[v], ew, denom, hbuf, acc + (size_t)v * NNODES * CDIM);
  }
  fuse_scores<<<(NNODES + 255) / 256, 256, 0, stream>>>(acc, b, fus_w, fus_b, wview);
  classify<<<(NNODES + 127) / 128, 128, 0, stream>>>(acc, b, w1, b1, w2, b2, wview, logits);
}
